// Model_91293824843960
// MI455X (gfx1250) — matrix operand order repaired and checked by reading, not compiled, not run on hardware
//
#include <hip/hip_runtime.h>
#include <hip/hip_bf16.h>
#include <math.h>

// Problem dims
#define BATCH 64
#define TSEQ  577
#define DDIM  768
#define SSEQ  577
// Everything padded to 608 = 19*32 rows/cols; D = 768 = 24*32.
#define PAD   608
#define NPADW 640           // weight rows padded so 64-wide col groups never guard loads
#define KCH   (DDIM / 32)   // 24 k-chunks
#define SCH   (PAD / 32)    // 19 s-chunks

typedef __bf16 bf16;
typedef __attribute__((ext_vector_type(16))) bf16  bf16x16;
typedef __attribute__((ext_vector_type(8)))  bf16  bf16x8;
typedef __attribute__((ext_vector_type(4)))  bf16  bf16x4;
typedef __attribute__((ext_vector_type(8)))  float f32x8;
typedef __attribute__((ext_vector_type(4)))  float f32x4;

static __device__ __forceinline__ int imin(int a, int b) { return a < b ? a : b; }

static __device__ __forceinline__ bf16x16 pack16(f32x4 a, f32x4 b, f32x4 c, f32x4 d) {
    bf16x16 r;
    r[0]=(bf16)a[0];  r[1]=(bf16)a[1];  r[2]=(bf16)a[2];  r[3]=(bf16)a[3];
    r[4]=(bf16)b[0];  r[5]=(bf16)b[1];  r[6]=(bf16)b[2];  r[7]=(bf16)b[3];
    r[8]=(bf16)c[0];  r[9]=(bf16)c[1];  r[10]=(bf16)c[2]; r[11]=(bf16)c[3];
    r[12]=(bf16)d[0]; r[13]=(bf16)d[1]; r[14]=(bf16)d[2]; r[15]=(bf16)d[3];
    return r;
}

static __device__ __forceinline__ f32x8 wmma_bf16(bf16x16 a, bf16x16 b, f32x8 c) {
    return __builtin_amdgcn_wmma_f32_16x16x32_bf16(
        false, a, false, b, (short)0, c, false, false);
}

// A fragment (16x32 bf16) from an f32 row: lane l15 = row,
// half selects K sub-chunk {half*16..+15}.
static __device__ __forceinline__ bf16x16 loadA_f32(const float* xrow, int k0, int half) {
    const f32x4* p0 = (const f32x4*)(xrow + k0 + half * 16);
    const f32x4* p1 = (const f32x4*)(xrow + k0 + half * 16 + 8);
    return pack16(p0[0], p0[1], p1[0], p1[1]);
}

// -----------------------------------------------------------------------------
// Prologue: convert weights f32 -> bf16, zero-padded to NPADW rows; pad bias
// (f32) with zeros. Grid: NPADW blocks of 192 threads (4 cols each).
// -----------------------------------------------------------------------------
__global__ __launch_bounds__(192)
void cvt_weight_bf16(const float* __restrict__ W, const float* __restrict__ bias,
                     bf16* __restrict__ Wb, float* __restrict__ biasb, int Nvalid)
{
    const int row = blockIdx.x;
    const int c0  = threadIdx.x * 4;
    bf16x4 o;
    if (row < Nvalid) {
        const f32x4 v = *(const f32x4*)(W + (size_t)row * DDIM + c0);
        o[0] = (bf16)v[0]; o[1] = (bf16)v[1]; o[2] = (bf16)v[2]; o[3] = (bf16)v[3];
    } else {
        o[0] = (bf16)0.0f; o[1] = (bf16)0.0f; o[2] = (bf16)0.0f; o[3] = (bf16)0.0f;
    }
    *(bf16x4*)(Wb + (size_t)row * DDIM + c0) = o;
    if (threadIdx.x == 0)
        biasb[row] = (row < Nvalid) ? bias[row] : 0.0f;
}

// -----------------------------------------------------------------------------
// Stage 1: Y[bi, m, n] = sum_k X[bi, m, k] * Wb[n, k] + biasb[n]  (bf16 out)
// Fully branch-free hot loop: A rows clamped to last valid row (pad rows only
// need finite values); Wb/biasb pre-padded with zeros so pad columns are 0.
// One wave computes a 32x64 tile: 8 accumulators, B fragments reused 2x,
// A fragments reused 4x. Grid: (PAD/32, NP/64 rounded up, BATCH), block 32.
// -----------------------------------------------------------------------------
__global__ __launch_bounds__(32)
void gemm_xwT_bias_bf16(const float* __restrict__ X,
                        const bf16*  __restrict__ Wb,
                        const float* __restrict__ biasb,
                        bf16* __restrict__ Y,
                        int NP)
{
    const int lane = threadIdx.x & 31;
    const int half = lane >> 4;
    const int l15  = lane & 15;
    const int bi   = blockIdx.z;
    const int m0   = blockIdx.x * 32;
    const int n0   = blockIdx.y * 64;

    const int ar0 = imin(m0 + l15,      TSEQ - 1);
    const int ar1 = imin(m0 + 16 + l15, TSEQ - 1);
    const float* xrow0 = X + ((size_t)bi * TSEQ + (size_t)ar0) * DDIM;
    const float* xrow1 = X + ((size_t)bi * TSEQ + (size_t)ar1) * DDIM;

    const bf16* wrow[4];
    for (int j = 0; j < 4; ++j)
        wrow[j] = Wb + (size_t)(n0 + j * 16 + l15) * DDIM;   // rows < NP rounded up to 64

    f32x8 acc[8];
    for (int j = 0; j < 8; ++j)
        for (int i = 0; i < 8; ++i) acc[j][i] = 0.0f;

    for (int k0 = 0; k0 < DDIM; k0 += 32) {
        __builtin_prefetch(xrow0 + k0 + 256, 0, 1);
        __builtin_prefetch(xrow1 + k0 + 256, 0, 1);
        const bf16x16 a0 = loadA_f32(xrow0, k0, half);
        const bf16x16 a1 = loadA_f32(xrow1, k0, half);
        const int kb = k0 + half * 16;   // B fragment: 16 contiguous K per lane
        for (int j = 0; j < 4; ++j) {
            const bf16x16 bfrag = *(const bf16x16*)(wrow[j] + kb);
            acc[j * 2 + 0] = wmma_bf16(a0, bfrag, acc[j * 2 + 0]);
            acc[j * 2 + 1] = wmma_bf16(a1, bfrag, acc[j * 2 + 1]);
        }
    }

    // C layout: VGPR i -> row (group*16 + half*8 + i), col = n
    for (int j = 0; j < 4; ++j) {
        const int n = n0 + j * 16 + l15;
        if (n < NP) {
            const float bsv = biasb[n];
            for (int g = 0; g < 2; ++g) {
                bf16* ybase = Y + ((size_t)bi * PAD + (size_t)(m0 + g * 16 + half * 8)) * NP + n;
                const f32x8 a = acc[j * 2 + g];
                for (int i = 0; i < 8; ++i)
                    ybase[(size_t)i * NP] = (bf16)(a[i] + bsv);
            }
        }
    }
}

// -----------------------------------------------------------------------------
// Stage 2 (fused): out[bi,t,u] = sum_s tanh( sum_d q[t,d] k[s,d] ) * v[u,s]
// One workgroup (4 waves) per (bi, 16-row t-tile). Q and P tiles live in LDS
// in WMMA A-fragment order [chunk][lane][16] -> every A operand is a single
// aligned 32-byte ds_load. Phases process PAIRS of 16-col tiles so each A
// fragment feeds two WMMAs.
// -----------------------------------------------------------------------------
__global__ __launch_bounds__(128)
void attn_fused(const bf16* __restrict__ Q,   // [BATCH, PAD, DDIM]
                const bf16* __restrict__ K,   // [BATCH, PAD, DDIM]
                const bf16* __restrict__ V,   // [BATCH, PAD, PAD]
                float* __restrict__ Out)      // [BATCH, TSEQ, TSEQ]
{
    __shared__ bf16x16 QsF[KCH * 32];   // 24 KB, fragment order
    __shared__ bf16x16 PsF[SCH * 32];   // 19 KB, fragment order

    const int tid  = threadIdx.x;
    const int lane = tid & 31;
    const int w    = tid >> 5;
    const int half = lane >> 4;
    const int l15  = lane & 15;
    const int bi   = blockIdx.y;
    const int t0   = blockIdx.x * 16;

    // Cooperative permuted load of the 16 x DDIM q tile into fragment order.
    {
        const bf16* Qbase = Q + ((size_t)bi * PAD + (size_t)t0) * DDIM;
        for (int idx = tid; idx < KCH * 64; idx += 128) {
            const int c   = idx >> 6;
            const int ln  = (idx >> 1) & 31;
            const int h2  = idx & 1;
            const int col = c * 32 + (ln >> 4) * 16 + h2 * 8;
            bf16x8* dst = (bf16x8*)(&QsF[c * 32 + ln]);
            dst[h2] = *(const bf16x8*)(Qbase + (size_t)(ln & 15) * DDIM + col);
        }
    }
    __syncthreads();

    // ---- Phase 1: P = tanh(Q K^T). 19 pairs of s-tiles split across 4 waves.
    for (int sp = w; sp < SCH; sp += 4) {
        const int s0 = sp * 32;
        f32x8 acc0, acc1;
        for (int i = 0; i < 8; ++i) { acc0[i] = 0.0f; acc1[i] = 0.0f; }
        const bf16* krow0 = K + ((size_t)bi * PAD + (size_t)(s0 + l15)) * DDIM;
        const bf16* krow1 = krow0 + (size_t)16 * DDIM;
        for (int c = 0; c < KCH; ++c) {
            const bf16x16 a  = QsF[c * 32 + lane];
            const int kb = c * 32 + half * 16;
            const bf16x16 b0 = *(const bf16x16*)(krow0 + kb);
            const bf16x16 b1 = *(const bf16x16*)(krow1 + kb);
            acc0 = wmma_bf16(a, b0, acc0);
            acc1 = wmma_bf16(a, b1, acc1);
        }
        // Scatter tanh(acc) into fragment-order PsF.
        // Value i of accG: t-row = half*8+i, s-col = s0 + G*16 + l15.
        bf16* P = (bf16*)(&PsF[0]);
        for (int gsel = 0; gsel < 2; ++gsel) {
            const f32x8 a = gsel ? acc1 : acc0;
            const int col = s0 + gsel * 16 + l15;
            const int c2     = col >> 5;
            const int r32    = col & 31;
            const int half_r = r32 >> 4;
            const int p      = r32 & 15;
            for (int i = 0; i < 8; ++i) {
                const int L = (half * 8 + i) + half_r * 16;   // reader lane
                P[((size_t)(c2 * 32 + L)) * 16 + p] = (bf16)tanhf(a[i]);
            }
        }
    }
    __syncthreads();

    // ---- Phase 2: out = P V^T. 19 pairs of u-tiles split across 4 waves.
    for (int up = w; up < SCH; up += 4) {
        const int u0 = up * 32;
        f32x8 acc0, acc1;
        for (int i = 0; i < 8; ++i) { acc0[i] = 0.0f; acc1[i] = 0.0f; }
        const bf16* vrow0 = V + ((size_t)bi * PAD + (size_t)(u0 + l15)) * PAD;
        const bf16* vrow1 = vrow0 + (size_t)16 * PAD;
        for (int c = 0; c < SCH; ++c) {
            const bf16x16 a  = PsF[c * 32 + lane];
            const int sb = c * 32 + half * 16;
            const bf16x16 b0 = *(const bf16x16*)(vrow0 + sb);
            const bf16x16 b1 = *(const bf16x16*)(vrow1 + sb);
            acc0 = wmma_bf16(a, b0, acc0);
            acc1 = wmma_bf16(a, b1, acc1);
        }
        const int uA = u0 + l15;
        const int uB = u0 + 16 + l15;
        for (int i = 0; i < 8; ++i) {
            const int t = t0 + half * 8 + i;
            if (t < TSEQ) {
                float* orow = Out + ((size_t)bi * TSEQ + (size_t)t) * TSEQ;
                if (uA < TSEQ) __builtin_nontemporal_store(acc0[i], orow + uA);
                if (uB < TSEQ) __builtin_nontemporal_store(acc1[i], orow + uB);
            }
        }
    }
}

// -----------------------------------------------------------------------------
// Launch
// -----------------------------------------------------------------------------
extern "C" void kernel_launch(void* const* d_in, const int* in_sizes, int n_in,
                              void* d_out, int out_size, void* d_ws, size_t ws_size,
                              hipStream_t stream)
{
    const float* b  = (const float*)d_in[0];
    const float* e  = (const float*)d_in[1];
    const float* Wq = (const float*)d_in[2];
    const float* bq = (const float*)d_in[3];
    const float* Wk = (const float*)d_in[4];
    const float* bk = (const float*)d_in[5];
    const float* Wv = (const float*)d_in[6];
    const float* bv = (const float*)d_in[7];
    float* out = (float*)d_out;

    // Workspace carve:
    //   q [B,608,768] bf16   k [B,608,768] bf16   v [B,608,608] bf16  (~167 MB)
    //   Wqb/Wkb/Wvb [640,768] bf16 (+~3 MB)   biasq/biask/biasv [640] f32
    bf16* qbf = (bf16*)d_ws;
    bf16* kbf = qbf + (size_t)BATCH * PAD * DDIM;
    bf16* vbf = kbf + (size_t)BATCH * PAD * DDIM;
    bf16* wqb = vbf + (size_t)BATCH * PAD * PAD;
    bf16* wkb = wqb + (size_t)NPADW * DDIM;
    bf16* wvb = wkb + (size_t)NPADW * DDIM;
    float* bqp = (float*)(wvb + (size_t)NPADW * DDIM);
    float* bkp = bqp + NPADW;
    float* bvp = bkp + NPADW;

    // Prologue: pack weights + biases (zero-padded).
    cvt_weight_bf16<<<NPADW, 192, 0, stream>>>(Wq, bq, wqb, bqp, DDIM);
    cvt_weight_bf16<<<NPADW, 192, 0, stream>>>(Wk, bk, wkb, bkp, DDIM);
    cvt_weight_bf16<<<NPADW, 192, 0, stream>>>(Wv, bv, wvb, bvp, SSEQ);

    const dim3 gQK(PAD / 32, DDIM / 64, BATCH);       // 19 x 12 x 64
    const dim3 gV (PAD / 32, (PAD + 63) / 64, BATCH); // 19 x 10 x 64

    gemm_xwT_bias_bf16<<<gQK, 32, 0, stream>>>(b, wqb, bqp, qbf, DDIM);
    gemm_xwT_bias_bf16<<<gQK, 32, 0, stream>>>(e, wkb, bkp, kbf, DDIM);
    gemm_xwT_bias_bf16<<<gV,  32, 0, stream>>>(b, wvb, bvp, vbf, PAD);

    attn_fused<<<dim3(PAD / 16, BATCH), 128, 0, stream>>>(qbf, kbf, vbf, out);
}
